// CRF_1529008357997
// MI455X (gfx1250) — compile-verified, matrix operand order checked by reading, not run on hardware
//
#include <hip/hip_runtime.h>
#include <hip/hip_bf16.h>

// CRF forward (negative log-likelihood), MI455X / gfx1250, wave32.
//
// Roofline: P stream = 67 MB (floor ~2.9us @ 23.3 TB/s; L2-resident on
// replay, 192MB L2). Scan is sequential over T=512, parallel only over
// B=256 -> 16 workgroups of 16 rows, 8 waves each (one 16-col output tile
// per wave). Per step per wave: 4x v_wmma_f32_16x16x32_bf16 (K=128) as two
// independent 2-deep accumulator chains. W = exp(A) stationary in registers.
// Z double-buffered in LDS (bf16) -> ONE barrier per step.
// P triple-buffered in LDS, streamed 2 steps ahead with
// global_load_async_to_lds_b128 (ASYNCcnt, in-order completion), waited
// with s_wait_asynccnt <= 2.

typedef __attribute__((ext_vector_type(16))) __bf16 v16bf;
typedef __attribute__((ext_vector_type(8)))  float  v8f;

#define BSZ     256
#define TLEN    512
#define NLAB    128
#define APITCH  130   // n+2 row stride of A
#define MTILE   16
#define PPITCH  132   // padded f32 pitch for staged P tile (bank-conflict free)
#define ZPITCH  136   // padded bf16 pitch for Z tile (multiple of 8 -> 16B chunks)

// Async global->LDS copy of 16 bytes (gfx1250 VGLOBAL async op, ASYNCcnt).
__device__ __forceinline__ void async_load_b128(unsigned lds_off, const void* gaddr) {
    asm volatile("global_load_async_to_lds_b128 %0, %1, off"
                 :: "v"(lds_off), "v"(gaddr)
                 : "memory");
}

__device__ __forceinline__ void wait_asynccnt0() {
#if __has_builtin(__builtin_amdgcn_s_wait_asynccnt)
    __builtin_amdgcn_s_wait_asynccnt(0);
#else
    asm volatile("s_wait_asynccnt 0x0" ::: "memory");
#endif
}

__device__ __forceinline__ void wait_asynccnt2() {
#if __has_builtin(__builtin_amdgcn_s_wait_asynccnt)
    __builtin_amdgcn_s_wait_asynccnt(2);
#else
    asm volatile("s_wait_asynccnt 0x2" ::: "memory");
#endif
}

__global__ __launch_bounds__(256)
void crf_fwd_kernel(const int* __restrict__ y,
                    const float* __restrict__ P,
                    const float* __restrict__ A,
                    float* __restrict__ out)
{
    __shared__ __align__(16) __bf16 Zlds[2][MTILE][ZPITCH];
    __shared__ __align__(16) float  Plds[3][MTILE][PPITCH];
    __shared__ float scoreLds[MTILE];
    __shared__ float rowAcc[MTILE];

    const int tid     = threadIdx.x;
    const int lane    = tid & 31;
    const int wv      = tid >> 5;          // wave id 0..7 -> output column tile
    const int rowBase = blockIdx.x * MTILE;

    if (tid < MTILE) { scoreLds[tid] = 0.f; rowAcc[tid] = 0.f; }
    __syncthreads();

    // ---------------- phase 0: score[b] = s1 (transitions) + s2 (emissions)
    {
        const int r  = tid >> 4;           // row within tile
        const int c  = tid & 15;           // chunk of 32 timesteps
        const int b  = rowBase + r;
        const int t0 = c * 32;
        const size_t pbase = (size_t)b * TLEN * NLAB;
        int yprev = (c == 0) ? NLAB : y[b * TLEN + t0 - 1];
        float s = 0.f;
        for (int tt = t0; tt < t0 + 32; ++tt) {
            const int yt = y[b * TLEN + tt];
            s += A[yprev * APITCH + yt];
            s += P[pbase + (size_t)tt * NLAB + yt];
            yprev = yt;
        }
        if (c == 15) s += A[yprev * APITCH + (NLAB + 1)];  // y_{T-1} -> end
        atomicAdd(&scoreLds[r], s);
    }

    // ---------------- phase 1: stationary W = exp(A) as bf16 B-operands
    // B-operand layout (16x16x32, 16-bit): lane<16 -> col N=lane, K = kb+{0..7},
    // kb+{16..23}; lane>=16 -> col N=lane-16, K = kb+{8..15}, kb+{24..31}.
    const int l15  = lane & 15;
    const int gcol = wv * 16 + l15;                 // global label column 0..127
    const int koff = (lane & 16) ? 8 : 0;
    v16bf bW[4];
    #pragma unroll
    for (int kt = 0; kt < 4; ++kt) {
        const int kb = kt * 32;
        #pragma unroll
        for (int q = 0; q < 8; ++q)
            bW[kt][q]     = (__bf16)__expf(A[(kb + koff + q) * APITCH + gcol]);
        #pragma unroll
        for (int q = 0; q < 8; ++q)
            bW[kt][8 + q] = (__bf16)__expf(A[(kb + 16 + koff + q) * APITCH + gcol]);
    }

    // ---------------- init Z0 = exp(A[n, :n]); async-stage P(0) and P(1)
    const int pm = tid >> 4;                        // staging row (0..15)
    const int pj = (tid & 15) * 8;                  // staging col base
    const size_t prow = (size_t)TLEN * NLAB;
    {
        #pragma unroll
        for (int q = 0; q < 8; ++q)
            Zlds[0][pm][pj + q] = (__bf16)__expf(A[NLAB * APITCH + pj + q]);
        const float* s0 = &P[(size_t)(rowBase + pm) * prow + pj];
        async_load_b128((unsigned)(uintptr_t)&Plds[0][pm][pj],     s0);
        async_load_b128((unsigned)(uintptr_t)&Plds[0][pm][pj + 4], s0 + 4);
        const float* s1 = s0 + NLAB;
        async_load_b128((unsigned)(uintptr_t)&Plds[1][pm][pj],     s1);
        async_load_b128((unsigned)(uintptr_t)&Plds[1][pm][pj + 4], s1 + 4);
    }
    wait_asynccnt2();   // P(0) landed (in-order); P(1) may be in flight
    __syncthreads();

    // ---------------- phase 2: forward scan, 512 chained steps, 1 barrier each
    const int mHalf = (lane >> 4) * 8;              // D-tile: vgpr r -> row mHalf+r
    float zfin[8];
    int zb = 0, pb = 0;
    #pragma unroll 2
    for (int t = 0; t < TLEN; ++t) {
        // fire-and-forget stage of P(t+2), two steps ahead (ASYNCcnt)
        if ((t + 2) < TLEN) {
            const int pn2 = (pb == 0) ? 2 : pb - 1;     // (pb + 2) % 3
            const float* src =
                &P[(size_t)(rowBase + pm) * prow + (size_t)(t + 2) * NLAB + pj];
            async_load_b128((unsigned)(uintptr_t)&Plds[pn2][pm][pj],     src);
            async_load_b128((unsigned)(uintptr_t)&Plds[pn2][pm][pj + 4], src + 4);
        }
        if (t + 16 < TLEN)
            __builtin_prefetch(
                &P[(size_t)(rowBase + pm) * prow + (size_t)(t + 16) * NLAB + pj], 0, 1);

        // A-operand: all of Z (16x128 bf16) from LDS buffer zb.
        // lane<16 -> row M=lane, K = kb+{0..7}, kb+{16..23} (two 16B chunks)
        v16bf aZ[4];
        #pragma unroll
        for (int kt = 0; kt < 4; ++kt) {
            const int kb = kt * 32;
            union { v16bf v; uint4 q[2]; } u;
            u.q[0] = *(const uint4*)&Zlds[zb][l15][kb + koff];
            u.q[1] = *(const uint4*)&Zlds[zb][l15][kb + 16 + koff];
            aZ[kt] = u.v;
        }

        // Znew tile = Z @ W : two independent 2-deep WMMA chains (K=64 each),
        // halving the matrix-pipe dependency depth per step.
        v8f acc0 = {}, acc1 = {};
        acc0 = __builtin_amdgcn_wmma_f32_16x16x32_bf16(false, aZ[0], false, bW[0],
                                                       (short)0, acc0, false, false);
        acc1 = __builtin_amdgcn_wmma_f32_16x16x32_bf16(false, aZ[2], false, bW[2],
                                                       (short)0, acc1, false, false);
        acc0 = __builtin_amdgcn_wmma_f32_16x16x32_bf16(false, aZ[1], false, bW[1],
                                                       (short)0, acc0, false, false);
        acc1 = __builtin_amdgcn_wmma_f32_16x16x32_bf16(false, aZ[3], false, bW[3],
                                                       (short)0, acc1, false, false);

        // elementwise * P_t; write Z_{t+1} into the OTHER Z buffer (no conflict
        // with Z_t readers), so a single barrier closes the step.
        #pragma unroll
        for (int r = 0; r < 8; ++r)
            zfin[r] = (acc0[r] + acc1[r]) * Plds[pb][mHalf + r][gcol];
        #pragma unroll
        for (int r = 0; r < 8; ++r)
            Zlds[zb ^ 1][mHalf + r][gcol] = (__bf16)zfin[r];

        wait_asynccnt2();   // P(t+1) landed (in-order); P(t+2) may be in flight
        __syncthreads();    // everyone's Z_{t+1} stores + P(t+1) visible
        zb ^= 1;
        pb = (pb == 2) ? 0 : pb + 1;
    }

    // ---------------- phase 3: logZ = log( sum_j Zf[:,j] * exp(A[j, n+1]) )
    const float stopf = __expf(A[gcol * APITCH + (NLAB + 1)]);
    #pragma unroll
    for (int r = 0; r < 8; ++r)
        atomicAdd(&rowAcc[mHalf + r], zfin[r] * stopf);
    __syncthreads();

    if (tid < MTILE)
        out[rowBase + tid] = __logf(rowAcc[tid]) - scoreLds[tid];
}

extern "C" void kernel_launch(void* const* d_in, const int* in_sizes, int n_in,
                              void* d_out, int out_size, void* d_ws, size_t ws_size,
                              hipStream_t stream) {
    const int*   y = (const int*)  d_in[0];   // (256, 512)
    const float* P = (const float*)d_in[1];   // (256, 512, 128)
    const float* A = (const float*)d_in[2];   // (130, 130)
    float* out = (float*)d_out;               // (256,)
    (void)in_sizes; (void)n_in; (void)out_size; (void)d_ws; (void)ws_size;

    dim3 grid(BSZ / MTILE);   // 16 workgroups, one 16-row tile each
    dim3 block(256);          // 8 wave32 -> 8 output column tiles
    hipLaunchKernelGGL(crf_fwd_kernel, grid, block, 0, stream, y, P, A, out);
}
